// LIIF_Generator_91302414778420
// MI455X (gfx1250) — matrix operand order repaired and checked by reading, not compiled, not run on hardware
//
#include <hip/hip_runtime.h>
#include <hip/hip_fp16.h>
#include <stdint.h>

typedef __attribute__((ext_vector_type(16))) _Float16 v16h;
typedef __attribute__((ext_vector_type(8)))  float    v8f;

#define HF 256
#define WF 256
#define KC 64
#define TSZ (9 * 64 * 64)   // one converted conv-weight tensor (halves)
#define XS 616              // X row stride (608 + 8 pad), halves
#define HS 264              // H row stride (256 + 8 pad), halves
#define PS 72               // conv window per-pixel stride (64 + 8 pad), halves

#define WMMA_F16(a, b, c) \
  __builtin_amdgcn_wmma_f32_16x16x32_f16(false, (a), false, (b), (short)0, (c), false, false)

// ---------------- CDNA5 async global->LDS staging (CDNA5 ISA, ASYNCcnt) ---------

__device__ __forceinline__ unsigned lds_off32(const void* p) {
  // flat shared-aperture address: low 32 bits are the LDS byte offset (ISA 10.2)
  return (unsigned)(uintptr_t)p;
}

__device__ __forceinline__ void async_ld_b128(const _Float16* lds_dst, const _Float16* gsrc) {
  asm volatile("global_load_async_to_lds_b128 %0, %1, off"
               :: "v"(lds_off32(lds_dst)),
                  "v"((unsigned long long)(uintptr_t)gsrc)
               : "memory");
}

__device__ __forceinline__ void async_wait0() {
  asm volatile("s_wait_asynccnt 0" ::: "memory");
}

// ---------------- WMMA fragment loaders (wave32 layouts per CDNA5 ISA) --------

// A (MxK, 16x32 f16): lanes 0-15 row M=lane hold K {0..7, 16..23}; lanes 16-31 row M=lane-16
// hold K {8..15, 24..31}.  Row-major source => two 16B loads per lane.
__device__ __forceinline__ v16h load_a_frag(const _Float16* base, int row0, int k0,
                                            int stride, int lane) {
  const int r    = lane & 15;
  const int koff = (lane < 16) ? 0 : 8;
  const _Float16* p = base + (row0 + r) * stride + k0 + koff;
  union { v16h v; uint4 q[2]; } u;
  u.q[0] = *(const uint4*)(p);
  u.q[1] = *(const uint4*)(p + 16);
  return u.v;
}

// B (KxN, 32x16 f16): lanes 0-15 column N=lane hold K {0..7, 16..23}; lanes 16-31 column N=lane-16
// hold K {8..15, 24..31}.  Weights stored [N][K] f16 => two 16B loads per lane.
__device__ __forceinline__ v16h load_b_frag(const _Float16* base, int n0, int k0,
                                            int stride, int lane) {
  const int n  = n0 + (lane & 15);
  const int kb = k0 + ((lane < 16) ? 0 : 8);
  const _Float16* p = base + (size_t)n * stride + kb;
  union { v16h v; uint4 q[2]; } u;
  u.q[0] = *(const uint4*)(p);
  u.q[1] = *(const uint4*)(p + 16);
  return u.v;
}

// ---------------- weight conversion kernels ----------------

struct ConvWPtrs { const float* p[17]; };

// f32 [Kout][Cin][3][3] -> f16 [tap r][n][c]  (B-operand layout, c contiguous)
__global__ void __launch_bounds__(256) conv_w_convert(ConvWPtrs ps, _Float16* __restrict__ dst) {
  const int idx = blockIdx.x * 256 + threadIdx.x;
  if (idx >= 17 * TSZ) return;
  const int t = idx / TSZ;
  const int e = idx - t * TSZ;
  const int r = e >> 12;
  const int n = (e >> 6) & 63;
  const int c = e & 63;
  dst[idx] = (_Float16)ps.p[t][(n * 64 + c) * 9 + r];
}

// l0w [580][256] -> B0 f16 [256][608]; input dims permuted c*9+r -> r*64+c; K zero-padded.
__global__ void __launch_bounds__(256) mlp_w0_convert(const float* __restrict__ w,
                                                      _Float16* __restrict__ B0) {
  const int idx = blockIdx.x * 256 + threadIdx.x;
  if (idx >= 256 * 608) return;
  const int n = idx / 608;
  const int k = idx - n * 608;
  float v = 0.0f;
  if (k < 576) {
    const int r = k >> 6, c = k & 63;
    v = w[(c * 9 + r) * 256 + n];
  } else if (k < 580) {
    v = w[k * 256 + n];          // rel0, rel1, cell0, cell1
  }
  B0[idx] = (_Float16)v;
}

// lw [256][256] -> f16 [256 out][256 in]
__global__ void __launch_bounds__(256) mlp_w_convert(const float* __restrict__ w,
                                                     _Float16* __restrict__ B) {
  const int idx = blockIdx.x * 256 + threadIdx.x;   // 65536 total
  const int n = idx >> 8, k = idx & 255;
  B[idx] = (_Float16)w[k * 256 + n];
}

__global__ void __launch_bounds__(256) mlp_w4_convert(const float* __restrict__ w,
                                                      _Float16* __restrict__ W4) {
  W4[threadIdx.x] = (_Float16)w[threadIdx.x];       // l4w [256][1]
}

// ---------------- conv0: 1 -> 64 channels (tiny, scalar) ----------------

__global__ void __launch_bounds__(256) conv0_simple(const float* __restrict__ lr,
                                                    const float* __restrict__ w0,
                                                    const float* __restrict__ b0,
                                                    _Float16* __restrict__ out) {
  const int idx = blockIdx.x * 256 + threadIdx.x;   // y*16384 + x*64 + k
  const int k = idx & 63;
  const int x = (idx >> 6) & 255;
  const int y = idx >> 14;
  float s = b0[k];
#pragma unroll
  for (int r = 0; r < 9; ++r) {
    const int yy = y + r / 3 - 1;
    const int xx = x + r % 3 - 1;
    if (yy >= 0 && yy < HF && xx >= 0 && xx < WF)
      s += w0[k * 9 + r] * lr[yy * WF + xx];
  }
  out[idx] = (_Float16)s;
}

// ---------------- WMMA 3x3 conv, 64->64 channels, f16 HWC activations ----------------
// Stage ONCE per block: 3x130-pixel input window (56KB, 72-half pixel stride => A-frag
// ds_load_b128 hits all 64 banks) + all 9 taps' weights (72KB), both via async-to-LDS.
// Then a barrier-free straight-line chain of 72 WMMAs per wave, operands pure LDS.

__global__ void __launch_bounds__(256) conv64_wmma(const _Float16* __restrict__ in,
                                                   const _Float16* __restrict__ wt,
                                                   const float* __restrict__ bias,
                                                   const _Float16* __restrict__ resid,
                                                   int do_relu,
                                                   _Float16* __restrict__ out) {
  __shared__ __align__(16) _Float16 Win[3 * 130 * PS];   // 56,160 B
  __shared__ __align__(16) _Float16 Wl[TSZ];             // 73,728 B
  const int y    = blockIdx.y;
  const int x0   = blockIdx.x * 128;
  const int tid  = threadIdx.x;
  const int lane = tid & 31;
  const int wv   = tid >> 5;

  // weights: 4608 x 16B async transfers
  for (int i = tid; i < TSZ / 8; i += 256)
    async_ld_b128(Wl + i * 8, wt + i * 8);

  // input window: pixels (rr,p) -> global (y+rr-1, x0+p-1); 8 x 16B chunks per pixel
  for (int i = tid; i < 3 * 130 * 8; i += 256) {
    const int pix  = i >> 3;
    const int part = i & 7;
    const int rr = pix / 130;
    const int p  = pix - rr * 130;
    const int yy = y + rr - 1;
    const int xx = x0 + p - 1;
    _Float16* dst = &Win[pix * PS + part * 8];
    if (yy < 0 || yy >= HF || xx < 0 || xx >= WF) {      // zero ('SAME') padding
      uint4 z; z.x = z.y = z.z = z.w = 0u;
      *(uint4*)dst = z;
    } else {
      async_ld_b128(dst, in + ((size_t)yy * WF + xx) * KC + part * 8);
    }
  }
  async_wait0();
  __syncthreads();

  v8f acc[4] = {};
#pragma unroll
  for (int r = 0; r < 9; ++r) {
    const int ki = r / 3, kj = r % 3;
    const _Float16* abase = Win + (ki * 130 + kj) * PS;  // col m => pixel m+kj (win col0 = x0-1)
    const _Float16* wtr   = Wl + r * (64 * 64);
#pragma unroll
    for (int kk = 0; kk < 2; ++kk) {
      v16h a  = load_a_frag(abase, wv * 16, kk * 32, PS, lane);
      v16h b0 = load_b_frag(wtr, 0,  kk * 32, 64, lane);
      v16h b1 = load_b_frag(wtr, 16, kk * 32, 64, lane);
      v16h b2 = load_b_frag(wtr, 32, kk * 32, 64, lane);
      v16h b3 = load_b_frag(wtr, 48, kk * 32, 64, lane);
      acc[0] = WMMA_F16(a, b0, acc[0]);
      acc[1] = WMMA_F16(a, b1, acc[1]);
      acc[2] = WMMA_F16(a, b2, acc[2]);
      acc[3] = WMMA_F16(a, b3, acc[3]);
    }
  }
  // epilogue: C layout -> lanes 0-15: N=lane, M=v; lanes 16-31: N=lane-16, M=8+v
  const int nl = lane & 15;
  const int mb = wv * 16 + ((lane < 16) ? 0 : 8);
#pragma unroll
  for (int nt = 0; nt < 4; ++nt) {
    const int n  = nt * 16 + nl;
    const float bv = bias[n];
#pragma unroll
    for (int v = 0; v < 8; ++v) {
      const size_t o = ((size_t)y * WF + (x0 + mb + v)) * KC + n;
      float val = acc[nt][v] + bv;
      if (resid)  val += (float)resid[o];
      if (do_relu) val = fmaxf(val, 0.0f);
      out[o] = (_Float16)val;
    }
  }
}

// ---------------- fused unfold-gather + 5-layer MLP + LIIF blend ----------------
// One workgroup = 64 queries.  Wave tile = 2 M-subtiles x 4 N-subtiles: per k-step
// 2 A-frags + 4 B-frags feed 8 WMMAs.  K-loop compile-time unrolled.

template <int KSTEPS>
__device__ __forceinline__ void gemm_layer(const _Float16* A, int astride,
                                           const _Float16* B, int bstride,
                                           const float* bias, _Float16* Hout,
                                           int wv, int lane) {
  const int m0 = (wv & 1) * 32;        // m-subtiles m0, m0+16
  const int n0 = (wv >> 1) * 64;       // n-subtiles n0 .. n0+48
  v8f acc[8] = {};
#pragma unroll
  for (int kk = 0; kk < KSTEPS; ++kk) {
    v16h a0 = load_a_frag(A, m0,      kk * 32, astride, lane);
    v16h a1 = load_a_frag(A, m0 + 16, kk * 32, astride, lane);
    v16h b0 = load_b_frag(B, n0,      kk * 32, bstride, lane);
    v16h b1 = load_b_frag(B, n0 + 16, kk * 32, bstride, lane);
    v16h b2 = load_b_frag(B, n0 + 32, kk * 32, bstride, lane);
    v16h b3 = load_b_frag(B, n0 + 48, kk * 32, bstride, lane);
    acc[0] = WMMA_F16(a0, b0, acc[0]);
    acc[1] = WMMA_F16(a0, b1, acc[1]);
    acc[2] = WMMA_F16(a0, b2, acc[2]);
    acc[3] = WMMA_F16(a0, b3, acc[3]);
    acc[4] = WMMA_F16(a1, b0, acc[4]);
    acc[5] = WMMA_F16(a1, b1, acc[5]);
    acc[6] = WMMA_F16(a1, b2, acc[6]);
    acc[7] = WMMA_F16(a1, b3, acc[7]);
  }
  const int nl    = lane & 15;
  const int rbase = (lane < 16) ? 0 : 8;
#pragma unroll
  for (int mi = 0; mi < 2; ++mi) {
#pragma unroll
    for (int ni = 0; ni < 4; ++ni) {
      const int n  = n0 + ni * 16 + nl;
      const float bv = bias[n];
      v8f c = acc[mi * 4 + ni];
#pragma unroll
      for (int v = 0; v < 8; ++v) {
        float val = fmaxf(c[v] + bv, 0.0f);            // bias + relu
        Hout[(m0 + mi * 16 + rbase + v) * HS + n] = (_Float16)val;
      }
    }
  }
  __syncthreads();
}

__global__ void __launch_bounds__(256) liif_mlp_wmma(
    const _Float16* __restrict__ feat, const float* __restrict__ loc,
    const float* __restrict__ cell,
    const _Float16* __restrict__ B0, const _Float16* __restrict__ B1,
    const _Float16* __restrict__ B2, const _Float16* __restrict__ B3,
    const float* __restrict__ bias0, const float* __restrict__ bias1,
    const float* __restrict__ bias2, const float* __restrict__ bias3,
    const _Float16* __restrict__ w4, const float* __restrict__ b4,
    float* __restrict__ out) {
  __shared__ __align__(16) _Float16 X [64 * XS];
  __shared__ __align__(16) _Float16 H0[64 * HS];
  __shared__ __align__(16) _Float16 H1[64 * HS];
  __shared__ float red[256];
  const int tid  = threadIdx.x;
  const int lane = tid & 31;
  const int wv   = tid >> 5;
  const int q0   = blockIdx.x * 64;

  // staging role: 4 threads per query row
  const int srow = tid >> 2;
  const int part = tid & 3;
  const int sq   = q0 + srow;
  const float sl0 = loc[(size_t)sq * 2 + 0];
  const float sl1 = loc[(size_t)sq * 2 + 1];
  const float sc0 = cell[(size_t)sq * 2 + 0] * 256.0f;
  const float sc1 = cell[(size_t)sq * 2 + 1] * 256.0f;

  // owner role: thread t<64 owns query q0+t; precompute LIIF areas for all 4 shifts
  float a0 = 0.f, a1 = 0.f, a2 = 0.f, a3 = 0.f, ret = 0.f;
  if (tid < 64) {
    const int oq = q0 + tid;
    const float L0 = loc[(size_t)oq * 2 + 0];
    const float L1 = loc[(size_t)oq * 2 + 1];
    for (int s = 0; s < 4; ++s) {
      const float vx = (s & 2) ? 1.f : -1.f;
      const float vy = (s & 1) ? 1.f : -1.f;
      float l0 = fminf(fmaxf(L0 + vx * (1.f / 256.f) + 1e-6f, -1.f + 1e-6f), 1.f - 1e-6f);
      float l1 = fminf(fmaxf(L1 + vy * (1.f / 256.f) + 1e-6f, -1.f + 1e-6f), 1.f - 1e-6f);
      int iy = (int)rintf((l0 + 1.f) * 128.f - 0.5f); iy = iy < 0 ? 0 : (iy > 255 ? 255 : iy);
      int ix = (int)rintf((l1 + 1.f) * 128.f - 0.5f); ix = ix < 0 ? 0 : (ix > 255 ? 255 : ix);
      const float rel0 = (L0 - (-1.f + (2.f * iy + 1.f) / 256.f)) * 256.f;
      const float rel1 = (L1 - (-1.f + (2.f * ix + 1.f) / 256.f)) * 256.f;
      const float ar = fabsf(rel0 * rel1) + 1e-9f;
      if (s == 0) a0 = ar; else if (s == 1) a1 = ar; else if (s == 2) a2 = ar; else a3 = ar;
    }
  }

  for (int s = 0; s < 4; ++s) {
    {   // ---- gather-stage X via async-to-LDS (reflect-pad 3x3 unfold, tap-major) ----
      const float vx = (s & 2) ? 1.f : -1.f;
      const float vy = (s & 1) ? 1.f : -1.f;
      float l0 = fminf(fmaxf(sl0 + vx * (1.f / 256.f) + 1e-6f, -1.f + 1e-6f), 1.f - 1e-6f);
      float l1 = fminf(fmaxf(sl1 + vy * (1.f / 256.f) + 1e-6f, -1.f + 1e-6f), 1.f - 1e-6f);
      int iy = (int)rintf((l0 + 1.f) * 128.f - 0.5f); iy = iy < 0 ? 0 : (iy > 255 ? 255 : iy);
      int ix = (int)rintf((l1 + 1.f) * 128.f - 0.5f); ix = ix < 0 ? 0 : (ix > 255 ? 255 : ix);
      for (int r = part; r < 9; r += 4) {
        int yy = iy + r / 3 - 1; yy = (yy < 0) ? -yy : ((yy > 255) ? 510 - yy : yy);
        int xx = ix + r % 3 - 1; xx = (xx < 0) ? -xx : ((xx > 255) ? 510 - xx : xx);
        const _Float16* src = feat + ((size_t)yy * WF + xx) * KC;
        _Float16* dst = &X[srow * XS + r * 64];
#pragma unroll
        for (int j = 0; j < 8; ++j) async_ld_b128(dst + j * 8, src + j * 8);
      }
      if (part == 3) {
        const float rel0 = (sl0 - (-1.f + (2.f * iy + 1.f) / 256.f)) * 256.f;
        const float rel1 = (sl1 - (-1.f + (2.f * ix + 1.f) / 256.f)) * 256.f;
        X[srow * XS + 576] = (_Float16)rel0;
        X[srow * XS + 577] = (_Float16)rel1;
        X[srow * XS + 578] = (_Float16)sc0;
        X[srow * XS + 579] = (_Float16)sc1;
        for (int j = 580; j < 608; ++j) X[srow * XS + j] = (_Float16)0.0f;
      }
    }
    async_wait0();
    __syncthreads();
    gemm_layer<19>(X,  XS, B0, 608, bias0, H0, wv, lane);
    gemm_layer< 8>(H0, HS, B1, 256, bias1, H1, wv, lane);
    gemm_layer< 8>(H1, HS, B2, 256, bias2, H0, wv, lane);
    gemm_layer< 8>(H0, HS, B3, 256, bias3, H1, wv, lane);
    // ---- layer 4 (256->1): 4 threads per query, 64 MACs each (vectorized), LDS reduce ----
    {
      const uint4* h4 = (const uint4*)(H1 + srow * HS + part * 64);
      const uint4* g4 = (const uint4*)(w4 + part * 64);
      float p = 0.f;
#pragma unroll
      for (int j = 0; j < 8; ++j) {
        union { uint4 q; _Float16 h[8]; } uh, uw;
        uh.q = h4[j]; uw.q = g4[j];
#pragma unroll
        for (int e = 0; e < 8; ++e) p += (float)uh.h[e] * (float)uw.h[e];
      }
      red[tid] = p;            // tid == srow*4 + part
    }
    __syncthreads();
    if (tid < 64) {
      float p = b4[0] + red[tid * 4 + 0] + red[tid * 4 + 1]
                      + red[tid * 4 + 2] + red[tid * 4 + 3];
      const float aw = (s == 0) ? a3 : (s == 1) ? a2 : (s == 2) ? a1 : a0;
      ret += p * aw;
    }
    __syncthreads();
  }
  if (tid < 64) out[q0 + tid] = ret / (a0 + a1 + a2 + a3);
}

// ---------------- host-side orchestration ----------------

extern "C" void kernel_launch(void* const* d_in, const int* in_sizes, int n_in,
                              void* d_out, int out_size, void* d_ws, size_t ws_size,
                              hipStream_t stream) {
  (void)in_sizes; (void)n_in; (void)out_size; (void)ws_size;
  const float* lr   = (const float*)d_in[0];
  const float* loc  = (const float*)d_in[1];
  const float* cell = (const float*)d_in[2];
  const float* w0   = (const float*)d_in[3];
  const float* b0   = (const float*)d_in[4];
  const float* wf   = (const float*)d_in[5];
  const float* bf   = (const float*)d_in[6];
  const float* rwa[8], *rba[8], *rwb[8], *rbb[8];
  for (int i = 0; i < 8; ++i) {
    rwa[i] = (const float*)d_in[7 + 4 * i];
    rba[i] = (const float*)d_in[8 + 4 * i];
    rwb[i] = (const float*)d_in[9 + 4 * i];
    rbb[i] = (const float*)d_in[10 + 4 * i];
  }
  const float* l0w = (const float*)d_in[39]; const float* l0b = (const float*)d_in[40];
  const float* l1w = (const float*)d_in[41]; const float* l1b = (const float*)d_in[42];
  const float* l2w = (const float*)d_in[43]; const float* l2b = (const float*)d_in[44];
  const float* l3w = (const float*)d_in[45]; const float* l3b = (const float*)d_in[46];
  const float* l4w = (const float*)d_in[47]; const float* l4b = (const float*)d_in[48];

  // workspace layout (f16 HWC activation buffers are 8MB each)
  char* ws = (char*)d_ws;
  const size_t ACT = (size_t)HF * WF * KC * sizeof(_Float16);   // 8 MB
  _Float16* act0 = (_Float16*)(ws);                // conv0 output / outer skip
  _Float16* actA = (_Float16*)(ws + ACT);          // residual-block temp
  _Float16* actB = (_Float16*)(ws + 2 * ACT);
  _Float16* actC = (_Float16*)(ws + 3 * ACT);
  _Float16* wtc  = (_Float16*)(ws + 4 * ACT);      // 17 conv weight tensors, f16
  _Float16* B0   = wtc + (size_t)17 * TSZ;
  _Float16* B1   = B0 + (size_t)256 * 608;
  _Float16* B2   = B1 + (size_t)256 * 256;
  _Float16* B3   = B2 + (size_t)256 * 256;
  _Float16* W4   = B3 + (size_t)256 * 256;

  // ---- weight conversion ----
  ConvWPtrs cp;
  for (int i = 0; i < 8; ++i) { cp.p[2 * i] = rwa[i]; cp.p[2 * i + 1] = rwb[i]; }
  cp.p[16] = wf;
  conv_w_convert<<<(17 * TSZ + 255) / 256, 256, 0, stream>>>(cp, wtc);
  mlp_w0_convert<<<(256 * 608 + 255) / 256, 256, 0, stream>>>(l0w, B0);
  mlp_w_convert<<<256, 256, 0, stream>>>(l1w, B1);
  mlp_w_convert<<<256, 256, 0, stream>>>(l2w, B2);
  mlp_w_convert<<<256, 256, 0, stream>>>(l3w, B3);
  mlp_w4_convert<<<1, 256, 0, stream>>>(l4w, W4);

  // ---- RDN backbone ----
  conv0_simple<<<(HF * WF * KC) / 256, 256, 0, stream>>>(lr, w0, b0, act0);

  dim3 cgrid(WF / 128, HF);
  _Float16* cur = act0;
  _Float16* outs[2] = { actB, actC };
  for (int i = 0; i < 8; ++i) {
    conv64_wmma<<<cgrid, 256, 0, stream>>>(cur, wtc + (size_t)(2 * i) * TSZ,
                                           rba[i], nullptr, 1, actA);
    _Float16* nxt = outs[i & 1];
    conv64_wmma<<<cgrid, 256, 0, stream>>>(actA, wtc + (size_t)(2 * i + 1) * TSZ,
                                           rbb[i], cur, 0, nxt);
    cur = nxt;
  }
  _Float16* feat = (cur == actB) ? actC : actB;
  conv64_wmma<<<cgrid, 256, 0, stream>>>(cur, wtc + (size_t)16 * TSZ, bf, act0, 0, feat);

  // ---- fused unfold + MLP + blend over the 512x512 query grid ----
  liif_mlp_wmma<<<(512 * 512) / 64, 256, 0, stream>>>(feat, loc, cell,
                                                      B0, B1, B2, B3,
                                                      l0b, l1b, l2b, l3b,
                                                      W4, l4b, (float*)d_out);
}
